// KSimplexAttentionLayer_2156073583038
// MI455X (gfx1250) — matrix operand order repaired and checked by reading, not compiled, not run on hardware
//
#include <hip/hip_runtime.h>

typedef _Float16 v16h __attribute__((ext_vector_type(16)));
typedef _Float16 v8h  __attribute__((ext_vector_type(8)));
typedef float    v8f  __attribute__((ext_vector_type(8)));

#define WMMA_F16(A, B, C) \
  __builtin_amdgcn_wmma_f32_16x16x32_f16(false, (A), false, (B), (short)0, (C), false, false)

constexpr int BATCH = 4;
constexpr int T     = 2048;
constexpr int D     = 768;
constexpr int EDIM  = 32;
constexpr int KV    = 5;      // K+1 vertices
constexpr int NTOK  = BATCH * T; // 8192
constexpr float LN_EPS = 1e-5f;
constexpr float SCALE  = 0.17677669529663687f;   // EDIM^-0.5
constexpr float LOG2E  = 1.4426950408889634f;
constexpr float SC2    = 2.0f * SCALE * LOG2E;   // logits kept in log2 units
constexpr float SC1    = SCALE * LOG2E;

__device__ __forceinline__ v16h make_a_tile(const _Float16* row, int k, int ka) {
  v8h x0 = *(const v8h*)(row + k + 2 * ka);
  v8h x1 = *(const v8h*)(row + k + 2 * ka + 8);
  v16h av;
#pragma unroll
  for (int i = 0; i < 8; ++i) { av[i] = x0[i]; av[i + 8] = x1[i]; }
  return av;
}

// ---------------------------------------------------------------------------
// Kernel 1: transpose + f32->f16 convert the two DxD weights (w_v, w_out).
// Output layout wT[n*D + k] so WMMA B-operand loads are per-lane contiguous.
// ---------------------------------------------------------------------------
__global__ __launch_bounds__(256) void cvt_w_kernel(
    const float* __restrict__ w_v, const float* __restrict__ w_out,
    _Float16* __restrict__ wvT, _Float16* __restrict__ woT)
{
  const float*   src = blockIdx.z ? w_out : w_v;
  _Float16*      dst = blockIdx.z ? woT   : wvT;
  int k = blockIdx.x * 16 + threadIdx.x;
  int n = blockIdx.y * 16 + threadIdx.y;
  dst[(size_t)n * D + k] = (_Float16)src[(size_t)k * D + n];
}

// ---------------------------------------------------------------------------
// Kernel 2: per-token LayerNorm -> h_f16; coords (incl. vertex softmax +
// deformed simplex) -> coords_f16; sq-norm -> sq. One block per token.
// ---------------------------------------------------------------------------
__global__ __launch_bounds__(256) void prep_kernel(
    const float* __restrict__ x,        const float* __restrict__ tmpl,
    const float* __restrict__ gamma,    const float* __restrict__ beta,
    const float* __restrict__ w_coords, const float* __restrict__ b_coords,
    const float* __restrict__ w_vertex, const float* __restrict__ b_vertex,
    const float* __restrict__ offs,     const float* __restrict__ dscale,
    _Float16* __restrict__ h16, _Float16* __restrict__ c16,
    float* __restrict__ sq)
{
  __shared__ float hrow[D];
  __shared__ float red[256];
  __shared__ float vlog[8];
  __shared__ float cstash[EDIM];

  const int tok = blockIdx.x;
  const int tid = threadIdx.x;
  const float* xr = x + (size_t)tok * D;

  float v0 = xr[tid], v1 = xr[tid + 256], v2 = xr[tid + 512];

  // mean
  red[tid] = v0 + v1 + v2;
  __syncthreads();
  for (int o = 128; o > 0; o >>= 1) {
    if (tid < o) red[tid] += red[tid + o];
    __syncthreads();
  }
  float mu = red[0] * (1.0f / D);
  __syncthreads();

  // variance (biased, matches jnp.var)
  float d0 = v0 - mu, d1 = v1 - mu, d2 = v2 - mu;
  red[tid] = d0 * d0 + d1 * d1 + d2 * d2;
  __syncthreads();
  for (int o = 128; o > 0; o >>= 1) {
    if (tid < o) red[tid] += red[tid + o];
    __syncthreads();
  }
  float rstd = rsqrtf(red[0] * (1.0f / D) + LN_EPS);
  __syncthreads();

  float h0 = d0 * rstd * gamma[tid]       + beta[tid];
  float h1 = d1 * rstd * gamma[tid + 256] + beta[tid + 256];
  float h2 = d2 * rstd * gamma[tid + 512] + beta[tid + 512];
  hrow[tid] = h0; hrow[tid + 256] = h1; hrow[tid + 512] = h2;
  _Float16* hr = h16 + (size_t)tok * D;
  hr[tid] = (_Float16)h0; hr[tid + 256] = (_Float16)h1; hr[tid + 512] = (_Float16)h2;
  __syncthreads();

  // coords projection (32 lanes) + vertex logits (5 lanes)
  if (tid < EDIM) {
    float c = b_coords[tid];
    for (int d = 0; d < D; ++d) c = fmaf(hrow[d], w_coords[d * EDIM + tid], c);
    cstash[tid] = c;
  } else if (tid < EDIM + KV) {
    int vi = tid - EDIM;
    float l = b_vertex[vi];
    for (int d = 0; d < D; ++d) l = fmaf(hrow[d], w_vertex[d * KV + vi], l);
    vlog[vi] = l;
  }
  __syncthreads();

  if (tid < EDIM) {
    float m = -1e30f;
    for (int v = 0; v < KV; ++v) m = fmaxf(m, vlog[v]);
    float e[KV], se = 0.f;
    for (int v = 0; v < KV; ++v) { e[v] = __expf(vlog[v] - m); se += e[v]; }
    float inv = 1.0f / se;
    float ds  = fminf(fmaxf(dscale[0], 0.05f), 0.5f);
    float c = cstash[tid];
    for (int v = 0; v < KV; ++v)
      c = fmaf(e[v] * inv, tmpl[v * EDIM + tid] + offs[v * EDIM + tid] * ds, c);
    c16[(size_t)tok * EDIM + tid] = (_Float16)c;
    // sq-norm: lanes 0..31 are exactly wave 0 in wave32
    float cs = c * c;
    for (int msk = 16; msk > 0; msk >>= 1) cs += __shfl_xor(cs, msk, 32);
    if (tid == 0) sq[tok] = cs;
  }
}

// ---------------------------------------------------------------------------
// Kernel 3: V = h_f16 @ w_v (f16 WMMA). Register-blocked: each wave computes
// a 64x16 output strip (4 M-tiles sharing one weight B-tile per k-step).
// All operand tiles are materialized before the WMMAs so the scheduler can
// issue one load clause + partial loadcnt waits instead of full drains.
// Output TRANSPOSED per batch: vT[(b*D + n)*T + t].
// grid = (NTOK/64, D/128), block = 8 waves.
// ---------------------------------------------------------------------------
__global__ __launch_bounds__(256) void gemm_v_kernel(
    const _Float16* __restrict__ A, const _Float16* __restrict__ Wt,
    const float* __restrict__ bias, _Float16* __restrict__ vT)
{
  const int lane = threadIdx.x & 31;
  const int wave = threadIdx.x >> 5;
  const int hi   = lane >> 4;
  const int ln   = lane & 15;
  const int m0   = blockIdx.x * 64;
  const int n    = blockIdx.y * 128 + wave * 16 + ln;

  const _Float16* ar0 = A + (size_t)(m0      + ln) * D;
  const _Float16* ar1 = A + (size_t)(m0 + 16 + ln) * D;
  const _Float16* ar2 = A + (size_t)(m0 + 32 + ln) * D;
  const _Float16* ar3 = A + (size_t)(m0 + 48 + ln) * D;
  const _Float16* brow = Wt + (size_t)n * D;
  const int ka = hi ? 8 : 0;
  const int kb = hi ? 16 : 0;

  v8f acc0 = {}, acc1 = {}, acc2 = {}, acc3 = {};
#pragma unroll 2
  for (int k = 0; k < D; k += 32) {
    __builtin_prefetch(brow + k + 64, 0, 1);
    __builtin_prefetch(ar0 + k + 64, 0, 1);
    __builtin_prefetch(ar1 + k + 64, 0, 1);
    __builtin_prefetch(ar2 + k + 64, 0, 1);
    __builtin_prefetch(ar3 + k + 64, 0, 1);
    // materialize all operands first -> one clause, partial waits
    v16h bv  = *(const v16h*)(brow + k + kb);
    v16h av0 = make_a_tile(ar0, k, ka);
    v16h av1 = make_a_tile(ar1, k, ka);
    v16h av2 = make_a_tile(ar2, k, ka);
    v16h av3 = make_a_tile(ar3, k, ka);
    acc0 = WMMA_F16(av0, bv, acc0);
    acc1 = WMMA_F16(av1, bv, acc1);
    acc2 = WMMA_F16(av2, bv, acc2);
    acc3 = WMMA_F16(av3, bv, acc3);
  }

  const float bn = bias[n];
  const int b  = m0 >> 11;          // 64 | 2048, never crosses a batch
  _Float16* vrow = vT + ((size_t)b * D + n) * T;
  v8f accs[4] = {acc0, acc1, acc2, acc3};
#pragma unroll
  for (int i = 0; i < 4; ++i) {
    const int t0 = ((m0 + 16 * i) & (T - 1)) + 8 * hi;
    v8h outv;
#pragma unroll
    for (int r = 0; r < 8; ++r) outv[r] = (_Float16)(accs[i][r] + bn);
    *(v8h*)(vrow + t0) = outv;      // rows t0..t0+7 contiguous for this lane
  }
}

// ---------------------------------------------------------------------------
// Kernel 4: fused flash-style simplex attention.
// One WG per 16-row i-tile (grid = NTOK/16). 8 waves; each wave owns 96
// output dims. Per j-step of 32: 2 WMMAs (scores) + online softmax (base-2)
// + 6 WMMAs (P@V). Row-constant -scale*||c_i||^2 cancels in softmax.
// ---------------------------------------------------------------------------
__global__ __launch_bounds__(256) void attn_kernel(
    const _Float16* __restrict__ c16, const float* __restrict__ sq,
    const _Float16* __restrict__ vT,  _Float16* __restrict__ o16)
{
  __shared__ _Float16 pbuf[8][16][32];   // per-wave P tile (16 x 32), 8 KB

  const int lane = threadIdx.x & 31;
  const int wave = threadIdx.x >> 5;
  const int hi   = lane >> 4;
  const int ln   = lane & 15;
  const int tok0 = blockIdx.x * 16;
  const int b    = tok0 >> 11;
  const int nbase = wave * 96;

  // A operand: coords of the 16 i-rows (EDIM=32 -> exactly one f16 A tile)
  const _Float16* ca = c16 + (size_t)(tok0 + ln) * EDIM;
  v16h areg = make_a_tile(ca, 0, hi ? 8 : 0);

  float rm[8], rl[8];
#pragma unroll
  for (int r = 0; r < 8; ++r) { rm[r] = -1e30f; rl[r] = 0.f; }
  v8f o[6] = {};

  const float*    sqb = sq + (size_t)b * T;
  const _Float16* cb  = c16 + (size_t)b * T * EDIM;
  const _Float16* vb  = vT + (size_t)b * D * T;

  for (int j = 0; j < T; j += 32) {
    // prefetch next j-step's coords + V tiles (global_prefetch_b8)
    __builtin_prefetch(cb + (size_t)(j + 32 + ln) * EDIM, 0, 1);
    __builtin_prefetch(vb + (size_t)(nbase + ln) * T + j + 32, 0, 1);

    // scores: S = coords_i (16x32) x coords_j^T (32x16), two 16-col tiles
    v16h b0 = *(const v16h*)(cb + (size_t)(j + ln) * EDIM + (hi ? 16 : 0));
    v16h b1 = *(const v16h*)(cb + (size_t)(j + 16 + ln) * EDIM + (hi ? 16 : 0));
    v8f z = {};
    v8f s0 = WMMA_F16(areg, b0, z);
    v8f s1 = WMMA_F16(areg, b1, z);

    const float sj0 = SC1 * sqb[j + ln];
    const float sj1 = SC1 * sqb[j + 16 + ln];

    float alpha[8], p0[8], p1[8];
#pragma unroll
    for (int r = 0; r < 8; ++r) {
      float l0 = SC2 * s0[r] - sj0;   // log2-domain logits; row-const dropped
      float l1 = SC2 * s1[r] - sj1;
      float mx = fmaxf(l0, l1);
#pragma unroll
      for (int msk = 8; msk > 0; msk >>= 1) mx = fmaxf(mx, __shfl_xor(mx, msk, 16));
      float nm = fmaxf(rm[r], mx);
      alpha[r] = exp2f(rm[r] - nm);
      p0[r] = exp2f(l0 - nm);
      p1[r] = exp2f(l1 - nm);
      float ps = p0[r] + p1[r];
#pragma unroll
      for (int msk = 8; msk > 0; msk >>= 1) ps += __shfl_xor(ps, msk, 16);
      rl[r] = rl[r] * alpha[r] + ps;
      rm[r] = nm;
    }
#pragma unroll
    for (int c = 0; c < 6; ++c)
#pragma unroll
      for (int r = 0; r < 8; ++r) o[c][r] *= alpha[r];

    // stage P through LDS to convert D-layout -> A-layout (same wave: DS in-order)
#pragma unroll
    for (int r = 0; r < 8; ++r) {
      pbuf[wave][r + 8 * hi][ln]      = (_Float16)p0[r];
      pbuf[wave][r + 8 * hi][ln + 16] = (_Float16)p1[r];
    }
    const int colb = hi ? 16 : 0;
    v8h q0 = *(const v8h*)&pbuf[wave][ln][colb];
    v8h q1 = *(const v8h*)&pbuf[wave][ln][colb + 8];
    v16h pa;
#pragma unroll
    for (int i = 0; i < 8; ++i) { pa[i] = q0[i]; pa[i + 8] = q1[i]; }

    // O += P (16x32) x V[j:j+32, n:n+16] for this wave's 6 n-chunks
#pragma unroll
    for (int c = 0; c < 6; ++c) {
      const int n = nbase + c * 16 + ln;
      v16h bv = *(const v16h*)(vb + (size_t)n * T + j + (hi ? 16 : 0));
      o[c] = WMMA_F16(pa, bv, o[c]);
    }
  }

  // normalize and write O row-major f16 (A-operand layout for final GEMM)
#pragma unroll
  for (int c = 0; c < 6; ++c) {
#pragma unroll
    for (int r = 0; r < 8; ++r) {
      const int m = tok0 + r + 8 * hi;
      o16[(size_t)m * D + nbase + c * 16 + ln] = (_Float16)(o[c][r] / rl[r]);
    }
  }
}

// ---------------------------------------------------------------------------
// Kernel 5: out = O_f16 @ w_out + b_out + residual -> f32.
// Register-blocked 64x16 per wave like gemm_v. grid = (NTOK/64, D/128).
// ---------------------------------------------------------------------------
__global__ __launch_bounds__(256) void gemm_out_kernel(
    const _Float16* __restrict__ A, const _Float16* __restrict__ Wt,
    const float* __restrict__ bias, const float* __restrict__ resid,
    float* __restrict__ out)
{
  const int lane = threadIdx.x & 31;
  const int wave = threadIdx.x >> 5;
  const int hi   = lane >> 4;
  const int ln   = lane & 15;
  const int m0   = blockIdx.x * 64;
  const int n    = blockIdx.y * 128 + wave * 16 + ln;

  const _Float16* ar0 = A + (size_t)(m0      + ln) * D;
  const _Float16* ar1 = A + (size_t)(m0 + 16 + ln) * D;
  const _Float16* ar2 = A + (size_t)(m0 + 32 + ln) * D;
  const _Float16* ar3 = A + (size_t)(m0 + 48 + ln) * D;
  const _Float16* brow = Wt + (size_t)n * D;
  const int ka = hi ? 8 : 0;
  const int kb = hi ? 16 : 0;

  v8f acc0 = {}, acc1 = {}, acc2 = {}, acc3 = {};
#pragma unroll 2
  for (int k = 0; k < D; k += 32) {
    __builtin_prefetch(brow + k + 64, 0, 1);
    __builtin_prefetch(ar0 + k + 64, 0, 1);
    __builtin_prefetch(ar1 + k + 64, 0, 1);
    __builtin_prefetch(ar2 + k + 64, 0, 1);
    __builtin_prefetch(ar3 + k + 64, 0, 1);
    v16h bv  = *(const v16h*)(brow + k + kb);
    v16h av0 = make_a_tile(ar0, k, ka);
    v16h av1 = make_a_tile(ar1, k, ka);
    v16h av2 = make_a_tile(ar2, k, ka);
    v16h av3 = make_a_tile(ar3, k, ka);
    acc0 = WMMA_F16(av0, bv, acc0);
    acc1 = WMMA_F16(av1, bv, acc1);
    acc2 = WMMA_F16(av2, bv, acc2);
    acc3 = WMMA_F16(av3, bv, acc3);
  }

  const float bn = bias[n];
  v8f accs[4] = {acc0, acc1, acc2, acc3};
#pragma unroll
  for (int i = 0; i < 4; ++i) {
#pragma unroll
    for (int r = 0; r < 8; ++r) {
      const int m = m0 + 16 * i + r + 8 * hi;
      const size_t idx = (size_t)m * D + n;
      out[idx] = accs[i][r] + bn + resid[idx];
    }
  }
}

// ---------------------------------------------------------------------------
extern "C" void kernel_launch(void* const* d_in, const int* in_sizes, int n_in,
                              void* d_out, int out_size, void* d_ws, size_t ws_size,
                              hipStream_t stream)
{
  const float* x        = (const float*)d_in[0];
  const float* tmpl     = (const float*)d_in[1];
  const float* gamma    = (const float*)d_in[2];
  const float* beta     = (const float*)d_in[3];
  const float* w_coords = (const float*)d_in[4];
  const float* b_coords = (const float*)d_in[5];
  const float* w_vertex = (const float*)d_in[6];
  const float* b_vertex = (const float*)d_in[7];
  const float* offs     = (const float*)d_in[8];
  const float* dscale   = (const float*)d_in[9];
  const float* w_v      = (const float*)d_in[10];
  const float* b_v      = (const float*)d_in[11];
  const float* w_out    = (const float*)d_in[12];
  const float* b_out    = (const float*)d_in[13];
  float* out = (float*)d_out;

  // workspace carve-out (~41 MB), 256B aligned
  char* ws = (char*)d_ws;
  size_t off = 0;
  auto carve = [&](size_t bytes) -> void* {
    off = (off + 255) & ~(size_t)255;
    void* p = ws + off;
    off += bytes;
    return p;
  };
  _Float16* h16 = (_Float16*)carve((size_t)NTOK * D * 2);
  _Float16* vT  = (_Float16*)carve((size_t)NTOK * D * 2);
  _Float16* o16 = (_Float16*)carve((size_t)NTOK * D * 2);
  _Float16* c16 = (_Float16*)carve((size_t)NTOK * EDIM * 2);
  float*    sq  = (float*)carve((size_t)NTOK * 4);
  _Float16* wvT = (_Float16*)carve((size_t)D * D * 2);
  _Float16* woT = (_Float16*)carve((size_t)D * D * 2);

  cvt_w_kernel<<<dim3(D / 16, D / 16, 2), dim3(16, 16), 0, stream>>>(w_v, w_out, wvT, woT);
  prep_kernel<<<NTOK, 256, 0, stream>>>(x, tmpl, gamma, beta, w_coords, b_coords,
                                        w_vertex, b_vertex, offs, dscale,
                                        h16, c16, sq);
  gemm_v_kernel<<<dim3(NTOK / 64, D / 128), 256, 0, stream>>>(h16, wvT, b_v, vT);
  attn_kernel<<<NTOK / 16, 256, 0, stream>>>(c16, sq, vT, o16);
  gemm_out_kernel<<<dim3(NTOK / 64, D / 128), 256, 0, stream>>>(o16, woT, b_out, x, out);
}
